// minCUTPooling_6777458393289
// MI455X (gfx1250) — matrix operand order repaired and checked by reading, not compiled, not run on hardware
//
#include <hip/hip_runtime.h>
#include <math.h>

#define GAT_N       50000
#define GAT_HEADS   4
#define GAT_HIDDEN  64
#define GAT_HD      256      // HEADS*HIDDEN
#define GAT_OUT     10
#define NEG_SLOPE   0.2f

typedef __attribute__((ext_vector_type(16))) _Float16 v16h;
typedef __attribute__((ext_vector_type(8)))  _Float16 v8h;
typedef __attribute__((ext_vector_type(8)))  float    v8f;

// ---------------------------------------------------------------------------
// f32 -> f16 conversion (activations), plain elementwise
// ---------------------------------------------------------------------------
__global__ void cvt_f32_to_f16(const float* __restrict__ in,
                               _Float16* __restrict__ out, long long n)
{
    long long i = (long long)blockIdx.x * blockDim.x + threadIdx.x;
    if (i < n) out[i] = (_Float16)in[i];
}

// ---------------------------------------------------------------------------
// Pack W [K, 256] f32 into WMMA B-fragment order, f16.
// Fragment (tileN, kb, lane) = 16 contiguous halves:
//   b[j] = W[(kb*32 + (lane>>4)*16 + j) * 256 + tileN*16 + (lane&15)]
// One thread per fragment (16 halves = 32B contiguous write).
// ---------------------------------------------------------------------------
__global__ void pack_B_f16(const float* __restrict__ W,
                           _Float16* __restrict__ Bpack, int K)
{
    int nKb = K >> 5;
    int idx = blockIdx.x * blockDim.x + threadIdx.x;     // (tileN*nKb + kb)*32 + lane
    if (idx >= 16 * nKb * 32) return;
    int lane = idx & 31;
    int t    = idx >> 5;
    int kb   = t % nKb;
    int tileN = t / nKb;
    int col = lane & 15;
    int kh  = lane >> 4;
    _Float16* dst = Bpack + (size_t)idx * 16;
    #pragma unroll
    for (int j = 0; j < 16; ++j)
        dst[j] = (_Float16)W[(size_t)(kb * 32 + kh * 16 + j) * GAT_HD + tileN * 16 + col];
}

// ---------------------------------------------------------------------------
// WMMA GEMM: C[M,256] = A[M,K] * B[K,256].
// A: row-major f16 (pre-converted). B: pre-packed f16 fragments.
// Each wave computes a 16x64 strip (4 accumulators), reusing the A fragment.
// Inner loop per k-block: 2x b128 A-loads, then 4x (2x b128 B-loads + WMMA).
// ---------------------------------------------------------------------------
__global__ __launch_bounds__(256) void gat_gemm_wmma(
    const _Float16* __restrict__ A, const _Float16* __restrict__ Bpack,
    float* __restrict__ C, int M, int K)
{
    const int lane  = threadIdx.x & 31;
    const int wave  = threadIdx.x >> 5;
    const int strip = blockIdx.x * 8 + wave;     // wave-uniform
    const int nStrips = ((M + 15) >> 4) * 4;     // 4 strips of 64 cols per row-tile
    if (strip >= nStrips) return;                // uniform: EXEC stays all-1s

    const int tileM  = strip >> 2;
    const int ntBase = (strip & 3) * 4;          // first of 4 N-tiles
    const int nKb    = K >> 5;

    const int r  = lane & 15;
    const int kh = lane >> 4;
    const _Float16* Arow = A + (size_t)(tileM * 16 + r) * K;

    v8f acc0 = {}, acc1 = {}, acc2 = {}, acc3 = {};

    for (int kb = 0; kb < nKb; ++kb) {
        const int k0 = kb * 32;
        // A 16x32 f16 fragment: one contiguous 16-half (32B) run per lane, k = kh*16 + j
        v8h alo = *(const v8h*)(Arow + k0 + kh * 16);
        v8h ahi = *(const v8h*)(Arow + k0 + kh * 16 + 8);
        v16h a;
        #pragma unroll
        for (int j = 0; j < 8; ++j) { a[j] = alo[j]; a[8 + j] = ahi[j]; }

        #pragma unroll
        for (int nt = 0; nt < 4; ++nt) {
            const _Float16* bp = Bpack +
                ((size_t)((ntBase + nt) * nKb + kb) * 32 + lane) * 16;
            v8h blo = *(const v8h*)(bp);
            v8h bhi = *(const v8h*)(bp + 8);
            v16h b;
            #pragma unroll
            for (int j = 0; j < 8; ++j) { b[j] = blo[j]; b[8 + j] = bhi[j]; }
            v8f& acc = (nt == 0) ? acc0 : (nt == 1) ? acc1 : (nt == 2) ? acc2 : acc3;
            acc = __builtin_amdgcn_wmma_f32_16x16x32_f16(
                false, a, false, b, (short)0, acc, false, false);
        }
    }

    // C/D f32 layout: col = lane&15, VGPR v -> row = v + 8*(lane>>4)
    const int col = lane & 15;
    float* Cbase = C + (size_t)(tileM * 16 + kh * 8) * GAT_HD + ntBase * 16 + col;
    #pragma unroll
    for (int v = 0; v < 8; ++v) {
        Cbase[(size_t)v * GAT_HD + 0]  = acc0[v];
        Cbase[(size_t)v * GAT_HD + 16] = acc1[v];
        Cbase[(size_t)v * GAT_HD + 32] = acc2[v];
        Cbase[(size_t)v * GAT_HD + 48] = acc3[v];
    }
}

// ---------------------------------------------------------------------------
// Attention logits: al_s[n,h] = <h[n,h,:], a_src[h,:]>, same for al_d
// ---------------------------------------------------------------------------
__global__ void gat_attn_logits(const float* __restrict__ h,
                                const float* __restrict__ a_src,
                                const float* __restrict__ a_dst,
                                float* __restrict__ al_s, float* __restrict__ al_d)
{
    int idx = blockIdx.x * blockDim.x + threadIdx.x;
    if (idx >= GAT_N * GAT_HEADS) return;
    int n  = idx / GAT_HEADS;
    int hd = idx % GAT_HEADS;
    const float* hp = h + (size_t)n * GAT_HD + hd * GAT_HIDDEN;
    const float* as = a_src + hd * GAT_HIDDEN;
    const float* ad = a_dst + hd * GAT_HIDDEN;
    float s = 0.f, d = 0.f;
    #pragma unroll 8
    for (int j = 0; j < GAT_HIDDEN; ++j) { float v = hp[j]; s += v * as[j]; d += v * ad[j]; }
    al_s[idx] = s; al_d[idx] = d;
}

// order-preserving float<->uint encoding so atomicMax(u32) == float max
__device__ __forceinline__ unsigned fenc(float f) {
    unsigned u = __float_as_uint(f);
    return (u & 0x80000000u) ? ~u : (u | 0x80000000u);
}
__device__ __forceinline__ float fdec(unsigned u) {
    return __uint_as_float((u & 0x80000000u) ? (u ^ 0x80000000u) : ~u);
}
#define FENC_NEG_INF 0x007FFFFFu   // fenc(-inf)

__global__ void fill_f32(float* p, float v, long long n) {
    long long i = (long long)blockIdx.x * blockDim.x + threadIdx.x;
    if (i < n) p[i] = v;
}
__global__ void fill_u32(unsigned* p, unsigned v, long long n) {
    long long i = (long long)blockIdx.x * blockDim.x + threadIdx.x;
    if (i < n) p[i] = v;
}

__device__ __forceinline__ void edge_sd(const int* __restrict__ src,
                                        const int* __restrict__ dst,
                                        long long e, int E0, int& s, int& d)
{
    if (e < E0) { s = src[e]; d = dst[e]; }
    else        { s = d = (int)(e - E0); }    // appended self-loops
}

__global__ void gat_edge_max(const int* __restrict__ src, const int* __restrict__ dst,
                             int E0, long long Etot,
                             const float* __restrict__ al_s, const float* __restrict__ al_d,
                             unsigned* __restrict__ mEnc)
{
    long long idx = (long long)blockIdx.x * blockDim.x + threadIdx.x;
    if (idx >= Etot * GAT_HEADS) return;
    int hd = (int)(idx % GAT_HEADS);
    long long e = idx / GAT_HEADS;
    int s, d; edge_sd(src, dst, e, E0, s, d);
    float ev = al_s[(size_t)s * GAT_HEADS + hd] + al_d[(size_t)d * GAT_HEADS + hd];
    ev = (ev > 0.f) ? ev : NEG_SLOPE * ev;    // leaky relu
    atomicMax(&mEnc[(size_t)d * GAT_HEADS + hd], fenc(ev));
}

__global__ void gat_edge_exp(const int* __restrict__ src, const int* __restrict__ dst,
                             int E0, long long Etot,
                             const float* __restrict__ al_s, const float* __restrict__ al_d,
                             const unsigned* __restrict__ mEnc,
                             float* __restrict__ exbuf, float* __restrict__ denom)
{
    long long idx = (long long)blockIdx.x * blockDim.x + threadIdx.x;
    if (idx >= Etot * GAT_HEADS) return;
    int hd = (int)(idx % GAT_HEADS);
    long long e = idx / GAT_HEADS;
    int s, d; edge_sd(src, dst, e, E0, s, d);
    float ev = al_s[(size_t)s * GAT_HEADS + hd] + al_d[(size_t)d * GAT_HEADS + hd];
    ev = (ev > 0.f) ? ev : NEG_SLOPE * ev;
    float m  = fdec(mEnc[(size_t)d * GAT_HEADS + hd]);
    float ex = __expf(ev - m);
    exbuf[idx] = ex;
    atomicAdd(&denom[(size_t)d * GAT_HEADS + hd], ex);
}

// weighted scatter-add: agg[dst,h,:] += alpha * h_lin[src,h,:]
// thread = (edge, head, half-row of 32); h_lin is L2-resident (51MB < 192MB L2)
__global__ void gat_edge_aggregate(const int* __restrict__ src, const int* __restrict__ dst,
                                   int E0, long long Etot,
                                   const float* __restrict__ exbuf,
                                   const float* __restrict__ denom,
                                   const float* __restrict__ hlin,
                                   float* __restrict__ agg)
{
    long long idx = (long long)blockIdx.x * blockDim.x + threadIdx.x;
    if (idx >= Etot * GAT_HEADS * 2) return;
    int half = (int)(idx & 1);
    long long t = idx >> 1;
    int hd = (int)(t % GAT_HEADS);
    long long e = t / GAT_HEADS;
    int s, d; edge_sd(src, dst, e, E0, s, d);
    float alpha = exbuf[e * GAT_HEADS + hd] / denom[(size_t)d * GAT_HEADS + hd];
    const float* hs = hlin + ((size_t)s * GAT_HEADS + hd) * GAT_HIDDEN + half * 32;
    float*       ag = agg  + ((size_t)d * GAT_HEADS + hd) * GAT_HIDDEN + half * 32;
    #pragma unroll
    for (int j = 0; j < 32; ++j) atomicAdd(&ag[j], alpha * hs[j]);
}

// head mean + bias + ELU -> next layer input [N, HIDDEN]
__global__ void gat_mean_elu(const float* __restrict__ agg,
                             const float* __restrict__ bias,
                             float* __restrict__ hout)
{
    long long idx = (long long)blockIdx.x * blockDim.x + threadIdx.x;
    if (idx >= (long long)GAT_N * GAT_HIDDEN) return;
    int n = (int)(idx / GAT_HIDDEN);
    int d = (int)(idx % GAT_HIDDEN);
    float s = 0.f;
    #pragma unroll
    for (int h = 0; h < GAT_HEADS; ++h)
        s += agg[((size_t)n * GAT_HEADS + h) * GAT_HIDDEN + d];
    s = s * (1.0f / GAT_HEADS) + bias[d];
    hout[idx] = (s > 0.f) ? s : (__expf(s) - 1.0f);   // ELU
}

// final FC: out[n,o] = <h[n,:], fcW[:,o]> + fcb[o]
__global__ void gat_fc(const float* __restrict__ h, const float* __restrict__ fcW,
                       const float* __restrict__ fcb, float* __restrict__ out)
{
    int idx = blockIdx.x * blockDim.x + threadIdx.x;
    if (idx >= GAT_N * GAT_OUT) return;
    int n = idx / GAT_OUT, o = idx % GAT_OUT;
    const float* hp = h + (size_t)n * GAT_HIDDEN;
    float s = fcb[o];
    #pragma unroll 8
    for (int j = 0; j < GAT_HIDDEN; ++j) s += hp[j] * fcW[(size_t)j * GAT_OUT + o];
    out[idx] = s;
}

// ---------------------------------------------------------------------------
extern "C" void kernel_launch(void* const* d_in, const int* in_sizes, int n_in,
                              void* d_out, int out_size, void* d_ws, size_t ws_size,
                              hipStream_t stream)
{
    const float* x   = (const float*)d_in[0];
    const int*   ei  = (const int*)d_in[1];
    const int    E0  = in_sizes[1] / 2;
    const int*   src = ei;
    const int*   dst = ei + E0;
    const float* W[3]    = {(const float*)d_in[2],  (const float*)d_in[6],  (const float*)d_in[10]};
    const float* asrc[3] = {(const float*)d_in[3],  (const float*)d_in[7],  (const float*)d_in[11]};
    const float* adst[3] = {(const float*)d_in[4],  (const float*)d_in[8],  (const float*)d_in[12]};
    const float* bias[3] = {(const float*)d_in[5],  (const float*)d_in[9],  (const float*)d_in[13]};
    const float* fcW = (const float*)d_in[14];
    const float* fcb = (const float*)d_in[15];
    float* out = (float*)d_out;

    const long long Etot = (long long)E0 + GAT_N;

    // workspace carve-up
    float*    h_lin = (float*)d_ws;                                  // N*256 f32
    float*    agg   = h_lin + (size_t)GAT_N * GAT_HD;                // N*256 f32
    float*    al_s  = agg   + (size_t)GAT_N * GAT_HD;                // N*4
    float*    al_d  = al_s  + (size_t)GAT_N * GAT_HEADS;             // N*4
    unsigned* mEnc  = (unsigned*)(al_d + (size_t)GAT_N * GAT_HEADS); // N*4
    float*    denom = (float*)(mEnc + (size_t)GAT_N * GAT_HEADS);    // N*4
    float*    exbuf = denom + (size_t)GAT_N * GAT_HEADS;             // Etot*4
    float*    hbuf  = exbuf + (size_t)Etot * GAT_HEADS;              // N*64 f32
    _Float16* Ahalf = (_Float16*)(hbuf + (size_t)GAT_N * GAT_HIDDEN); // N*128 f16 max
    _Float16* Bpack = Ahalf + (size_t)GAT_N * 128;                    // 16*(K/32)*32*16 f16

    const int  THR = 256;
    const long long nEH  = Etot * GAT_HEADS;
    const long long nAgg = (long long)GAT_N * GAT_HD;

    const float* cur = x;
    for (int layer = 0; layer < 3; ++layer) {
        const int K = (layer == 0) ? 128 : GAT_HIDDEN;

        // stage activations as f16, pack W as f16 fragments
        long long nA = (long long)GAT_N * K;
        cvt_f32_to_f16<<<(unsigned)((nA + THR - 1) / THR), THR, 0, stream>>>(cur, Ahalf, nA);
        int nFrag = 16 * (K >> 5) * 32;
        pack_B_f16<<<(nFrag + THR - 1) / THR, THR, 0, stream>>>(W[layer], Bpack, K);

        // h_lin = cur @ W  (WMMA, 16x64 strip per wave)
        int strips = ((GAT_N + 15) / 16) * 4;
        gat_gemm_wmma<<<(strips + 7) / 8, THR, 0, stream>>>(Ahalf, Bpack, h_lin, GAT_N, K);

        gat_attn_logits<<<(GAT_N * GAT_HEADS + THR - 1) / THR, THR, 0, stream>>>(
            h_lin, asrc[layer], adst[layer], al_s, al_d);

        fill_f32<<<(unsigned)((nAgg + THR - 1) / THR), THR, 0, stream>>>(agg, 0.f, nAgg);
        fill_f32<<<(GAT_N * GAT_HEADS + THR - 1) / THR, THR, 0, stream>>>(
            denom, 0.f, (long long)GAT_N * GAT_HEADS);
        fill_u32<<<(GAT_N * GAT_HEADS + THR - 1) / THR, THR, 0, stream>>>(
            mEnc, FENC_NEG_INF, (long long)GAT_N * GAT_HEADS);

        gat_edge_max<<<(unsigned)((nEH + THR - 1) / THR), THR, 0, stream>>>(
            src, dst, E0, Etot, al_s, al_d, mEnc);
        gat_edge_exp<<<(unsigned)((nEH + THR - 1) / THR), THR, 0, stream>>>(
            src, dst, E0, Etot, al_s, al_d, mEnc, exbuf, denom);
        gat_edge_aggregate<<<(unsigned)((nEH * 2 + THR - 1) / THR), THR, 0, stream>>>(
            src, dst, E0, Etot, exbuf, denom, h_lin, agg);

        gat_mean_elu<<<(unsigned)(((long long)GAT_N * GAT_HIDDEN + THR - 1) / THR), THR, 0, stream>>>(
            agg, bias[layer], hbuf);
        cur = hbuf;
    }

    gat_fc<<<(GAT_N * GAT_OUT + THR - 1) / THR, THR, 0, stream>>>(hbuf, fcW, fcb, out);
}
